// GatedSwitchGNN_11227044512149
// MI455X (gfx1250) — matrix operand order repaired and checked by reading, not compiled, not run on hardware
//
#include <hip/hip_runtime.h>

#define BN   200
#define VN   64
#define HD   64
#define FINN 32
#define NSW  10
#define EC   63
#define MN   73
#define VLOWF 0.9f
#define VUPPF 1.1f

typedef __attribute__((ext_vector_type(16))) _Float16 v16h;
typedef __attribute__((ext_vector_type(8)))  _Float16 v8h;
typedef __attribute__((ext_vector_type(8)))  float    v8f;

__device__ __forceinline__ float sigmoidf_(float x) { return 1.0f / (1.0f + __expf(-x)); }

// ---------------------------------------------------------------------------
// prep: conn = clip(A+S,0,1), dinv = 1/max(deg,1), T = embed_s @ p0_Ws (2xHD)
// ---------------------------------------------------------------------------
__global__ void prep_kernel(const float* __restrict__ A, const float* __restrict__ S,
                            const float* __restrict__ emb, const float* __restrict__ p0_Ws,
                            float* __restrict__ conn, float* __restrict__ dinv,
                            float* __restrict__ T) {
  __shared__ float cL[VN * VN];
  const int t = threadIdx.x;  // 128
  for (int i = t; i < VN * VN; i += 128) {
    float c = A[i] + S[i];
    c = c > 1.0f ? 1.0f : c;
    cL[i] = c;
    conn[i] = c;
  }
  __syncthreads();
  if (t < VN) {
    float s = 0.0f;
    for (int w = 0; w < VN; ++w) s += cL[t * VN + w];
    dinv[t] = 1.0f / fmaxf(s, 1.0f);
  }
  if (t < 2 * HD) {
    const int r = t / HD, h = t % HD;
    float s = 0.0f;
    for (int f = 0; f < FINN; ++f) s += emb[r * FINN + f] * p0_Ws[f * HD + h];
    T[t] = s;
  }
}

// ---------------------------------------------------------------------------
// node linears: xi=x@Wi, xj=x@Wj, yv=x@Vw, xu=x@U  via WMMA f16 (K = 32 or 64)
// block = 128 threads (4 waves); wave m handles matrix m; block = 16 rows of M
// ---------------------------------------------------------------------------
template <int K>
__global__ void node_linear_kernel(const float* __restrict__ x,
                                   const float* __restrict__ W0, const float* __restrict__ W1,
                                   const float* __restrict__ W2, const float* __restrict__ W3,
                                   float* __restrict__ o0, float* __restrict__ o1,
                                   float* __restrict__ o2, float* __restrict__ o3) {
  constexpr int NK = K / 32;
  const int wave = threadIdx.x >> 5;
  const int lane = threadIdx.x & 31;
  const int hl = lane >> 4;   // lane half selects K sub-chunk
  const int r = lane & 15;
  const int rowbase = blockIdx.x * 16;
  const float* W = (wave == 0) ? W0 : (wave == 1) ? W1 : (wave == 2) ? W2 : W3;
  float* O = (wave == 0) ? o0 : (wave == 1) ? o1 : (wave == 2) ? o2 : o3;

  // B fragments: element e -> (K = ks*32 + hl*8 + (e&7) + (e>>3)*16, N = nt*16 + r)
  v16h bf[4][NK];
  for (int nt = 0; nt < 4; ++nt)
    for (int ks = 0; ks < NK; ++ks)
      for (int e = 0; e < 16; ++e)
        bf[nt][ks][e] = (_Float16)W[(ks * 32 + hl * 8 + (e & 7) + ((e >> 3) << 4)) * HD + nt * 16 + r];

  // A fragments: row = rowbase + r; vectorized float4 loads, cvt to f16
  const float* xr = x + (size_t)(rowbase + r) * K;
  v16h af[NK];
  for (int ks = 0; ks < NK; ++ks) {
    const float4 a0 = *(const float4*)(xr + ks * 32 + hl * 8);
    const float4 a1 = *(const float4*)(xr + ks * 32 + hl * 8 + 4);
    const float4 b0 = *(const float4*)(xr + ks * 32 + 16 + hl * 8);
    const float4 b1 = *(const float4*)(xr + ks * 32 + 16 + hl * 8 + 4);
    v16h a;
    a[0] = (_Float16)a0.x;  a[1] = (_Float16)a0.y;  a[2] = (_Float16)a0.z;  a[3] = (_Float16)a0.w;
    a[4] = (_Float16)a1.x;  a[5] = (_Float16)a1.y;  a[6] = (_Float16)a1.z;  a[7] = (_Float16)a1.w;
    a[8] = (_Float16)b0.x;  a[9] = (_Float16)b0.y;  a[10] = (_Float16)b0.z; a[11] = (_Float16)b0.w;
    a[12] = (_Float16)b1.x; a[13] = (_Float16)b1.y; a[14] = (_Float16)b1.z; a[15] = (_Float16)b1.w;
    af[ks] = a;
  }

  v8f acc[4];
  for (int nt = 0; nt < 4; ++nt)
    for (int k = 0; k < 8; ++k) acc[nt][k] = 0.0f;
  for (int nt = 0; nt < 4; ++nt)
    for (int ks = 0; ks < NK; ++ks)
      acc[nt] = __builtin_amdgcn_wmma_f32_16x16x32_f16(false, af[ks], false, bf[nt][ks],
                                                       (short)0, acc[nt], false, false);
  // C/D layout: (vgpr k, lane) -> M = k + 8*hl, N = r
  for (int nt = 0; nt < 4; ++nt)
    for (int k = 0; k < 8; ++k)
      O[(size_t)(rowbase + k + 8 * hl) * HD + nt * 16 + r] = acc[nt][k];
}

// ---------------------------------------------------------------------------
// fused edge layer: one workgroup = (b,v); 4 waves x 16 w-rows
//   e = s@Ws (+T lookup for layer 0) + xi[v] + xj[w] ; e = relu(LN(e))
//   s_out = s + e (residual except first); gate = sigmoid(s_out)*conn
//   msg[h] = dinv[v] * sum_w gate[w,h]*yv[w,h]
//   x_next = x + relu(LN(xu + msg))   (no residual on first layer)
// s is touched exactly once (read) + once (write) per layer: residual comes
// from the sprevL LDS cache filled during the WMMA A-fragment load.
// ---------------------------------------------------------------------------
template <bool FIRST>
__global__ void edge_layer_kernel(_Float16* __restrict__ s_io,
                                  const float* __restrict__ xi, const float* __restrict__ xj,
                                  const float* __restrict__ yv, const float* __restrict__ xu,
                                  const float* __restrict__ Ws_g,  // HDxHD (unused if FIRST)
                                  const float* __restrict__ T,     // 2xHD  (FIRST only)
                                  const float* __restrict__ S,     // VNxVN (FIRST only)
                                  const float* __restrict__ conn, const float* __restrict__ dinv,
                                  const float* __restrict__ x_cur,  // unused if FIRST
                                  float* __restrict__ x_next) {
  __shared__ _Float16 WsL[HD * HD];     // 8 KB
  __shared__ _Float16 sprevL[VN * HD];  // 8 KB: s values cached for residual
  __shared__ float yL[VN * 68];         // 17.4 KB, padded (float4-aligned, conflict-free cols)
  __shared__ float ebuf[VN * 65];       // 16.6 KB, e tile then gate buffer
  __shared__ float xiL[HD];
  __shared__ float connL[VN];
  __shared__ int sidxL[VN];
  __shared__ float Tl[2 * HD];
  __shared__ float msgP[2 * HD];
  __shared__ float ubuf[HD];

  const int t = threadIdx.x;  // 128
  const int b = blockIdx.x / VN;
  const int v = blockIdx.x % VN;
  const size_t nodeb = (size_t)(b * VN) * HD;
  const size_t srow0 = ((size_t)(b * VN + v)) * VN * HD;

  if (!FIRST) {
    for (int iv = t; iv < (HD * HD) / 4; iv += 128) {
      const float4 wv = ((const float4*)Ws_g)[iv];
      WsL[iv * 4 + 0] = (_Float16)wv.x;
      WsL[iv * 4 + 1] = (_Float16)wv.y;
      WsL[iv * 4 + 2] = (_Float16)wv.z;
      WsL[iv * 4 + 3] = (_Float16)wv.w;
    }
  }
  for (int iv = t; iv < (VN * HD) / 4; iv += 128) {
    const int w = iv >> 4, c4 = (iv & 15) * 4;
    const float4 yv4 = *(const float4*)(yv + nodeb + (size_t)w * HD + c4);
    *(float4*)&yL[w * 68 + c4] = yv4;
  }
  if (t < HD) xiL[t] = xi[nodeb + (size_t)v * HD + t];
  if (t < VN) connL[t] = conn[v * VN + t];
  if (FIRST) {
    if (t < VN) sidxL[t] = (S[v * VN + t] > 0.5f) ? 1 : 0;
    if (t < 2 * HD) Tl[t] = T[t];
  }
  __syncthreads();

  if (FIRST) {
    // layer-0 s@Ws has only 2 distinct rows -> table lookup
    for (int i = t; i < VN * HD; i += 128) {
      const int w = i >> 6, h = i & 63;
      ebuf[w * 65 + h] = Tl[sidxL[w] * HD + h] + xiL[h] + xj[nodeb + i];
    }
  } else {
    const int wave = t >> 5, lane = t & 31, hl = lane >> 4, r = lane & 15;
    v16h bf[4][2];
    for (int nt = 0; nt < 4; ++nt)
      for (int ks = 0; ks < 2; ++ks)
        for (int e = 0; e < 16; ++e)
          bf[nt][ks][e] = WsL[(ks * 32 + hl * 8 + (e & 7) + ((e >> 3) << 4)) * HD + nt * 16 + r];

    const int arow = wave * 16 + r;
    const _Float16* sp = s_io + srow0 + (size_t)arow * HD;
    v16h af[2];
    for (int ks = 0; ks < 2; ++ks) {
      union { v16h v; v8h h2[2]; } u;
      u.h2[0] = *(const v8h*)(sp + ks * 32 + hl * 8);
      u.h2[1] = *(const v8h*)(sp + ks * 32 + 16 + hl * 8);
      // cache for the residual pass (avoids a second 105MB global read)
      *(v8h*)&sprevL[arow * HD + ks * 32 + hl * 8] = u.h2[0];
      *(v8h*)&sprevL[arow * HD + ks * 32 + 16 + hl * 8] = u.h2[1];
      af[ks] = u.v;
    }
    v8f acc[4];
    for (int nt = 0; nt < 4; ++nt)
      for (int k = 0; k < 8; ++k) acc[nt][k] = 0.0f;
    for (int nt = 0; nt < 4; ++nt)
      for (int ks = 0; ks < 2; ++ks)
        acc[nt] = __builtin_amdgcn_wmma_f32_16x16x32_f16(false, af[ks], false, bf[nt][ks],
                                                         (short)0, acc[nt], false, false);
    for (int nt = 0; nt < 4; ++nt)
      for (int k = 0; k < 8; ++k) {
        const int wl = wave * 16 + k + 8 * hl;
        const int h = nt * 16 + r;
        ebuf[wl * 65 + h] = acc[nt][k] + xiL[h] + xj[nodeb + (size_t)wl * HD + h];
      }
  }
  __syncthreads();

  // per-row LN + relu + residual + gate; 2 threads per w-row, paired via shfl
  {
    const int row = t >> 1, part = t & 1;
    float s1 = 0.0f, s2 = 0.0f;
    for (int i = 0; i < 32; ++i) {
      const float e = ebuf[row * 65 + part * 32 + i];
      s1 += e; s2 += e * e;
    }
    s1 += __shfl_xor(s1, 1, 32);
    s2 += __shfl_xor(s2, 1, 32);
    const float m = s1 * (1.0f / HD);
    const float var = s2 * (1.0f / HD) - m * m;
    const float rs = rsqrtf(var + 1e-5f);
    _Float16* srow = s_io + srow0 + (size_t)row * HD;
    const float cg = connL[row];
    v8h svec[4];
    for (int i = 0; i < 32; ++i) {
      const int h = part * 32 + i;
      float e = (ebuf[row * 65 + h] - m) * rs;
      e = fmaxf(e, 0.0f);
      const float so = FIRST ? e : e + (float)sprevL[row * HD + h];
      svec[i >> 3][i & 7] = (_Float16)so;
      ebuf[row * 65 + h] = sigmoidf_(so) * cg;  // reuse as gate buffer
    }
    for (int q = 0; q < 4; ++q)
      *(v8h*)(srow + part * 32 + q * 8) = svec[q];  // vectorized 16B f16 stores
  }
  __syncthreads();

  // msg[h] = dinv[v] * sum_w gate[w,h] * y[w,h]  (two partial sums over w)
  {
    const int h = t & 63, seg = t >> 6;
    float p = 0.0f;
    for (int w = seg * 32; w < seg * 32 + 32; ++w) p += ebuf[w * 65 + h] * yL[w * 68 + h];
    msgP[seg * HD + h] = p;
  }
  __syncthreads();
  if (t < HD) {
    const float msg = (msgP[t] + msgP[HD + t]) * dinv[v];
    ubuf[t] = xu[nodeb + (size_t)v * HD + t] + msg;
  }
  __syncthreads();
  if (t < HD) {
    float s1 = 0.0f, s2 = 0.0f;
    for (int h = 0; h < HD; ++h) {
      const float u = ubuf[h];
      s1 += u; s2 += u * u;
    }
    const float m = s1 * (1.0f / HD);
    const float var = s2 * (1.0f / HD) - m * m;
    const float hv = fmaxf((ubuf[t] - m) * rsqrtf(var + 1e-5f), 0.0f);
    const size_t idx = nodeb + (size_t)v * HD + t;
    x_next[idx] = FIRST ? hv : x_cur[idx] + hv;
  }
}

// ---------------------------------------------------------------------------
// x_g[b,h] = sum_v xh[b,v,h]
// ---------------------------------------------------------------------------
__global__ void xg_kernel(const float* __restrict__ xh, float* __restrict__ xg) {
  const int b = blockIdx.x, h = threadIdx.x;
  float s = 0.0f;
  for (int v = 0; v < VN; ++v) s += xh[((size_t)b * VN + v) * HD + h];
  xg[b * HD + h] = s;
}

// ---------------------------------------------------------------------------
// switch head via WMMA: 16 gathered rows/block; in(256) @ W1(256x256) -> relu
// -> @ W2(256x4) -> sigmoid. 128 threads: 4 waves x 4 n-tiles, 8 k-steps.
// ---------------------------------------------------------------------------
__global__ void smlp_wmma_kernel(const _Float16* __restrict__ s_h, const float* __restrict__ xh,
                                 const float* __restrict__ xg, const int* __restrict__ edge_S,
                                 const float* __restrict__ W1, const float* __restrict__ W2,
                                 float* __restrict__ outp) {
  constexpr int KD = 4 * HD;  // 256
  __shared__ _Float16 inL[16 * KD];
  __shared__ float y1[16 * (KD + 4)];
  const int t = threadIdx.x;  // 128
  const int row0 = blockIdx.x * 16;
  __builtin_prefetch(W2, 0, 3);
  for (int i = t; i < 16 * KD; i += 128) {
    const int rr = i >> 8, c = i & 255;
    int grow = row0 + rr;
    if (grow >= BN * NSW) grow = BN * NSW - 1;
    const int b = grow / NSW, e = grow % NSW;
    const int si = edge_S[e], sj = edge_S[NSW + e];
    _Float16 val;
    if (c < HD)          val = s_h[(((size_t)(b * VN + si)) * VN + sj) * HD + c];
    else if (c < 2 * HD) val = (_Float16)xh[((size_t)b * VN + si) * HD + (c - HD)];
    else if (c < 3 * HD) val = (_Float16)xh[((size_t)b * VN + sj) * HD + (c - 2 * HD)];
    else                 val = (_Float16)xg[b * HD + (c - 3 * HD)];
    inL[i] = val;
  }
  __syncthreads();
  const int wave = t >> 5, lane = t & 31, hl = lane >> 4, r = lane & 15;
  v8f acc[4];
  for (int q = 0; q < 4; ++q)
    for (int k = 0; k < 8; ++k) acc[q][k] = 0.0f;
  for (int ks = 0; ks < KD / 32; ++ks) {
    union { v16h v; v8h h2[2]; } u;
    u.h2[0] = *(const v8h*)&inL[r * KD + ks * 32 + hl * 8];
    u.h2[1] = *(const v8h*)&inL[r * KD + ks * 32 + 16 + hl * 8];
    for (int q = 0; q < 4; ++q) {
      const int nt = wave * 4 + q;
      v16h bfr;
      for (int e2 = 0; e2 < 16; ++e2)
        bfr[e2] = (_Float16)W1[(ks * 32 + hl * 8 + (e2 & 7) + ((e2 >> 3) << 4)) * KD + nt * 16 + r];
      acc[q] = __builtin_amdgcn_wmma_f32_16x16x32_f16(false, u.v, false, bfr,
                                                      (short)0, acc[q], false, false);
    }
  }
  for (int q = 0; q < 4; ++q) {
    const int nt = wave * 4 + q;
    for (int k = 0; k < 8; ++k)
      y1[(k + 8 * hl) * (KD + 4) + nt * 16 + r] = fmaxf(acc[q][k], 0.0f);
  }
  __syncthreads();
  if (t < 64) {
    const int rr = t >> 2, c = t & 3;
    const int grow = row0 + rr;
    if (grow < BN * NSW) {
      float o = 0.0f;
      for (int k = 0; k < KD; ++k) o += y1[rr * (KD + 4) + k] * W2[k * 4 + c];
      outp[grow * 4 + c] = sigmoidf_(o);
    }
  }
}

// ---------------------------------------------------------------------------
// connection head via WMMA: in(192) @ W1(192x192) -> relu -> @ W2(192x3)
// 128 threads: 4 waves x 3 n-tiles, 6 k-steps.
// ---------------------------------------------------------------------------
__global__ void cmlp_wmma_kernel(const float* __restrict__ xh, const float* __restrict__ xg,
                                 const int* __restrict__ edge_A, const float* __restrict__ W1,
                                 const float* __restrict__ W2, float* __restrict__ outp) {
  constexpr int KD = 3 * HD;  // 192
  __shared__ _Float16 inL[16 * KD];
  __shared__ float y1[16 * (KD + 4)];
  const int t = threadIdx.x;  // 128
  const int row0 = blockIdx.x * 16;
  __builtin_prefetch(W2, 0, 3);
  for (int i = t; i < 16 * KD; i += 128) {
    const int rr = i / KD, c = i % KD;
    int grow = row0 + rr;
    if (grow >= BN * EC) grow = BN * EC - 1;
    const int b = grow / EC, e = grow % EC;
    float val;
    if (c < HD)          val = xh[((size_t)b * VN + edge_A[e]) * HD + c];
    else if (c < 2 * HD) val = xh[((size_t)b * VN + edge_A[EC + e]) * HD + (c - HD)];
    else                 val = xg[b * HD + (c - 2 * HD)];
    inL[i] = (_Float16)val;
  }
  __syncthreads();
  const int wave = t >> 5, lane = t & 31, hl = lane >> 4, r = lane & 15;
  v8f acc[3];
  for (int q = 0; q < 3; ++q)
    for (int k = 0; k < 8; ++k) acc[q][k] = 0.0f;
  for (int ks = 0; ks < KD / 32; ++ks) {
    union { v16h v; v8h h2[2]; } u;
    u.h2[0] = *(const v8h*)&inL[r * KD + ks * 32 + hl * 8];
    u.h2[1] = *(const v8h*)&inL[r * KD + ks * 32 + 16 + hl * 8];
    for (int q = 0; q < 3; ++q) {
      const int nt = wave * 3 + q;
      v16h bfr;
      for (int e2 = 0; e2 < 16; ++e2)
        bfr[e2] = (_Float16)W1[(ks * 32 + hl * 8 + (e2 & 7) + ((e2 >> 3) << 4)) * KD + nt * 16 + r];
      acc[q] = __builtin_amdgcn_wmma_f32_16x16x32_f16(false, u.v, false, bfr,
                                                      (short)0, acc[q], false, false);
    }
  }
  for (int q = 0; q < 3; ++q) {
    const int nt = wave * 3 + q;
    for (int k = 0; k < 8; ++k)
      y1[(k + 8 * hl) * (KD + 4) + nt * 16 + r] = fmaxf(acc[q][k], 0.0f);
  }
  __syncthreads();
  if (t < 48) {
    const int rr = t / 3, c = t % 3;
    const int grow = row0 + rr;
    if (grow < BN * EC) {
      float o = 0.0f;
      for (int k = 0; k < KD; ++k) o += y1[rr * (KD + 4) + k] * W2[k * 3 + c];
      outp[grow * 3 + c] = sigmoidf_(o);
    }
  }
}

// ---------------------------------------------------------------------------
// final assembly: out[b] = [p_flow(73), v(64), graph_topo(73)]
// ---------------------------------------------------------------------------
__global__ void final_kernel(const float* __restrict__ smlp_out, const float* __restrict__ cmlp_out,
                             const float* __restrict__ Dinv, const float* __restrict__ Incp,
                             const float* __restrict__ Incc, float* __restrict__ out) {
  const int b = blockIdx.x, t = threadIdx.x;  // 128
  float* ob = out + (size_t)b * (2 * MN + VN);
  if (t < MN) {
    float p, gt;
    if (t < EC) {
      p = cmlp_out[(b * EC + t) * 3 + 0] - 0.5f;
      gt = 1.0f;
    } else {
      const float* so = smlp_out + (b * NSW + (t - EC)) * 4;
      p = so[1] - 0.5f;
      gt = so[0];
    }
    ob[t] = p;
    ob[MN + VN + t] = gt;
  }
  if (t >= 64 && t < 128) {
    const int vi = t - 64;
    const float d = Dinv[vi * VN + vi];
    float acc = 0.0f;
    for (int m2 = 0; m2 < MN; ++m2) {
      float vp, vc;
      if (m2 < EC) {
        const float* co = cmlp_out + (b * EC + m2) * 3;
        vp = VLOWF + (VUPPF - VLOWF) * co[1];
        vc = VLOWF + (VUPPF - VLOWF) * co[2];
      } else {
        const float* so = smlp_out + (b * NSW + (m2 - EC)) * 4;
        vp = VLOWF + (VUPPF - VLOWF) * so[2];
        vc = VLOWF + (VUPPF - VLOWF) * so[3];
      }
      acc += vp * Incp[vi * MN + m2] + vc * Incc[vi * MN + m2];
    }
    float val = d * acc;
    if (vi == 0) val = 1.0f;
    ob[MN + vi] = val;
  }
}

// ---------------------------------------------------------------------------
extern "C" void kernel_launch(void* const* d_in, const int* in_sizes, int n_in,
                              void* d_out, int out_size, void* d_ws, size_t ws_size,
                              hipStream_t stream) {
  (void)in_sizes; (void)n_in; (void)out_size; (void)ws_size;
  const float* x     = (const float*)d_in[0];
  const float* A     = (const float*)d_in[1];
  const float* S     = (const float*)d_in[2];
  const float* emb   = (const float*)d_in[3];
  const float* p0_Ws = (const float*)d_in[4];
  const float* p0_Wi = (const float*)d_in[5];
  const float* p0_Wj = (const float*)d_in[6];
  const float* p0_U  = (const float*)d_in[7];
  const float* p0_V  = (const float*)d_in[8];
  const float* pl_Ws = (const float*)d_in[9];
  const float* pl_Wi = (const float*)d_in[10];
  const float* pl_Wj = (const float*)d_in[11];
  const float* pl_U  = (const float*)d_in[12];
  const float* pl_V  = (const float*)d_in[13];
  const float* smW1  = (const float*)d_in[14];
  const float* smW2  = (const float*)d_in[15];
  const float* cmW1  = (const float*)d_in[16];
  const float* cmW2  = (const float*)d_in[17];
  const float* Dinv  = (const float*)d_in[18];
  const float* Incp  = (const float*)d_in[19];
  const float* Incc  = (const float*)d_in[20];
  const int* edge_A  = (const int*)d_in[21];
  const int* edge_S  = (const int*)d_in[22];
  float* outp = (float*)d_out;

  char* ws = (char*)d_ws;
  size_t off = 0;
  auto alloc = [&](size_t bytes) -> char* {
    char* p = ws + off;
    off = (off + bytes + 255) & ~(size_t)255;
    return p;
  };
  const size_t nodeN = (size_t)BN * VN * HD;  // 819200
  _Float16* s_h = (_Float16*)alloc((size_t)BN * VN * VN * HD * sizeof(_Float16));
  float* xb0 = (float*)alloc(nodeN * 4);
  float* xb1 = (float*)alloc(nodeN * 4);
  float* xi  = (float*)alloc(nodeN * 4);
  float* xj  = (float*)alloc(nodeN * 4);
  float* yv  = (float*)alloc(nodeN * 4);
  float* xu  = (float*)alloc(nodeN * 4);
  float* Tt  = (float*)alloc(2 * HD * 4);
  float* conn = (float*)alloc(VN * VN * 4);
  float* dinv = (float*)alloc(VN * 4);
  float* xg  = (float*)alloc((size_t)BN * HD * 4);
  float* smo = (float*)alloc((size_t)BN * NSW * 4 * 4);
  float* cmo = (float*)alloc((size_t)BN * EC * 3 * 4);

  prep_kernel<<<1, 128, 0, stream>>>(A, S, emb, p0_Ws, conn, dinv, Tt);

  // layer 0 (K = FIN = 32; edge GEMM replaced by 2-row table T)
  node_linear_kernel<FINN><<<(BN * VN) / 16, 128, 0, stream>>>(x, p0_Wi, p0_Wj, p0_V, p0_U,
                                                               xi, xj, yv, xu);
  edge_layer_kernel<true><<<BN * VN, 128, 0, stream>>>(s_h, xi, xj, yv, xu, nullptr, Tt, S,
                                                       conn, dinv, nullptr, xb0);
  // layer 1
  node_linear_kernel<HD><<<(BN * VN) / 16, 128, 0, stream>>>(xb0, pl_Wi, pl_Wj, pl_V, pl_U,
                                                             xi, xj, yv, xu);
  edge_layer_kernel<false><<<BN * VN, 128, 0, stream>>>(s_h, xi, xj, yv, xu, pl_Ws, nullptr,
                                                        nullptr, conn, dinv, xb0, xb1);
  // layer 2
  node_linear_kernel<HD><<<(BN * VN) / 16, 128, 0, stream>>>(xb1, pl_Wi + HD * HD,
                                                             pl_Wj + HD * HD, pl_V + HD * HD,
                                                             pl_U + HD * HD, xi, xj, yv, xu);
  edge_layer_kernel<false><<<BN * VN, 128, 0, stream>>>(s_h, xi, xj, yv, xu, pl_Ws + HD * HD,
                                                        nullptr, nullptr, conn, dinv, xb1, xb0);
  // heads
  xg_kernel<<<BN, HD, 0, stream>>>(xb0, xg);
  smlp_wmma_kernel<<<(BN * NSW) / 16, 128, 0, stream>>>(s_h, xb0, xg, edge_S, smW1, smW2, smo);
  cmlp_wmma_kernel<<<(BN * EC + 15) / 16, 128, 0, stream>>>(xb0, xg, edge_A, cmW1, cmW2, cmo);
  final_kernel<<<BN, 128, 0, stream>>>(smo, cmo, Dinv, Incp, Incc, outp);
}
